// HybridAttentionQKVPacked_64261300683344
// MI455X (gfx1250) — matrix operand order repaired and checked by reading, not compiled, not run on hardware
//
#include <hip/hip_runtime.h>

typedef __attribute__((ext_vector_type(16))) _Float16 v16h;
typedef __attribute__((ext_vector_type(8)))  float    v8f;
typedef unsigned int u32x4 __attribute__((ext_vector_type(4)));
typedef int          i32x4 __attribute__((ext_vector_type(4)));
typedef int          i32x8 __attribute__((ext_vector_type(8)));

namespace {
constexpr int Bc = 2, Sc = 2048, Hc = 32, Dc = 128;
constexpr int NW = 8;            // waves per block
constexpr int QT = 16;           // query rows per wave
constexpr int KC = 32;           // keys per chunk (WMMA K)
constexpr int BQ = NW * QT;      // 128 query rows per block
constexpr int STRIDE_S = 3 * Hc * Dc;  // floats per sequence step in packed qkv
}

__device__ __forceinline__ v8f vzero8() {
  v8f z;
#pragma unroll
  for (int i = 0; i < 8; ++i) z[i] = 0.0f;
  return z;
}

// One XOR-butterfly step for all 8 rows at once: 8 independent ds_swizzle ops
// (single wait), then 8 independent v_max.
template <int IMM>
__device__ __forceinline__ void swz_max8(float (&mx)[8]) {
  float t[8];
#pragma unroll
  for (int r = 0; r < 8; ++r)
    t[r] = __int_as_float(
        __builtin_amdgcn_ds_swizzle(__float_as_int(mx[r]), IMM));
#pragma unroll
  for (int r = 0; r < 8; ++r) mx[r] = fmaxf(mx[r], t[r]);
}

__global__ __launch_bounds__(NW * 32) void fa_fwd_f16wmma(const float* __restrict__ qkv,
                                                          float* __restrict__ out) {
  __shared__ float    Stg[KC][2][Dc];      // 32 KB f32 staging (TDM target)
  __shared__ _Float16 KVs[2][KC][Dc];      // 16 KB f16 K(0)/V(1) tiles
  __shared__ _Float16 Ps[NW][QT][KC];      // 8 KB per-wave P relayout buffers

  const int tid  = threadIdx.x;
  const int wave = tid >> 5;
  const int lane = tid & 31;
  const int ln   = lane & 15;
  const int hi   = lane >> 4;

  const int nqb = Sc / BQ;                 // 16 query-blocks per (b,h)
  const int bh  = blockIdx.x / nqb;
  const int qb  = blockIdx.x % nqb;
  const int b   = bh / Hc;
  const int h   = bh % Hc;
  const int q_base = qb * BQ;
  const int q0 = q_base + wave * QT;       // this wave's first query row

  const float* qptr = qkv + (size_t)b * Sc * STRIDE_S + (size_t)h * Dc;
  const float* kptr = qptr + Hc * Dc;      // qkv[..,1,..]; V is +H*D inside tile

  // LDS byte offset of the staging buffer (addrspace(3) ptrtoint)
  const unsigned stg_off =
      (unsigned)(uintptr_t)(__attribute__((address_space(3))) float*)&Stg[0][0][0];

  // ---- TDM descriptor: loop-invariant groups, built once ----
  // 3D tile: [x=128 f32][y=2 (K,V planes, stride H*D)][z=32 rows, stride 3*H*D]
  i32x8 g1c;
  g1c[0] = (int)(2u << 16);                // data_size=2 (4B), workgroup_mask=0
  g1c[1] = (int)(128u << 16);              // tensor_dim0=128 (bits 63:48)
  g1c[2] = (int)(2u << 16);                // dim0 hi=0 | tensor_dim1=2
  g1c[3] = (int)(128u << 16);              // dim1 hi=0 | tile_dim0=128
  g1c[4] = (int)((32u << 16) | 2u);        // tile_dim1=2 | tile_dim2=32
  g1c[5] = 4096;                           // tensor_dim0_stride = H*D
  g1c[6] = (int)(12288u << 16);            // tensor_dim1_stride[15:0] = 3*H*D
  g1c[7] = 0;
  i32x4 g2c; g2c[0] = 1 << 20; g2c[1] = 0; g2c[2] = 0; g2c[3] = 0;
  i32x4 g3c; g3c[0] = 0; g3c[1] = 0; g3c[2] = 0; g3c[3] = 0;
#if defined(__clang_major__) && __clang_major__ >= 23
  i32x8 g4c;
#pragma unroll
  for (int i = 0; i < 8; ++i) g4c[i] = 0;
#endif
  auto tdm_issue = [&](const float* krow) {
    const unsigned long long ga = (unsigned long long)(uintptr_t)krow;
    u32x4 g0;
    g0[0] = 1u;                                    // count=1 (valid)
    g0[1] = stg_off;                               // lds_addr (bytes)
    g0[2] = (unsigned)ga;                          // global_addr[31:0]
    g0[3] = (unsigned)((ga >> 32) & 0x1FFFFFFull)  // global_addr[56:32]
            | (2u << 30);                          // type=2 ("image")
#if defined(__clang_major__) && __clang_major__ >= 23
    __builtin_amdgcn_tensor_load_to_lds(g0, g1c, g2c, g3c, g4c, 0);
#else
    __builtin_amdgcn_tensor_load_to_lds(g0, g1c, g2c, g3c, 0);
#endif
  };

  // ---- load Q tile (16x128) into WMMA A-layout registers ----
  // Scale folds in 1/sqrt(128) AND log2(e): softmax runs in exp2 domain.
  constexpr float scale = 0.08838834764831845f * 1.44269504088896340736f;
  v16h qa[4];
  {
    const float* qr = qptr + (size_t)(q0 + ln) * STRIDE_S;
#pragma unroll
    for (int c = 0; c < 4; ++c) {
      const int d0 = c * 32 + hi * 8;
#pragma unroll
      for (int e = 0; e < 8; ++e) {
        qa[c][e]     = (_Float16)(qr[d0 + e] * scale);
        qa[c][8 + e] = (_Float16)(qr[d0 + 16 + e] * scale);
      }
    }
  }

  // all-ones B matrix: row-sum of P via one WMMA
  v16h vones;
#pragma unroll
  for (int e = 0; e < 16; ++e) vones[e] = (_Float16)1.0f;

  v8f oacc[8];
#pragma unroll
  for (int t = 0; t < 8; ++t) oacc[t] = vzero8();
  float mrow[8], lrow[8];
#pragma unroll
  for (int r = 0; r < 8; ++r) { mrow[r] = -1e30f; lrow[r] = 0.0f; }

  // B-operand builders (LDS -> registers, used by the pipelined GEMM loops)
  auto ldb = [&](const _Float16* kr, int c) {
    v16h bk;
    const int d0 = c * 32 + hi * 16;
#pragma unroll
    for (int e = 0; e < 16; ++e) bk[e] = kr[d0 - hi * 8 + e + (e >= 8 ? 8 : 0)];
    return bk;
  };
  auto ldv = [&](int dt) {
    v16h bv;
    const int dcol = dt * 16 + ln;
#pragma unroll
    for (int e = 0; e < 16; ++e) bv[e] = KVs[1][hi * 8 + e + (e >= 8 ? 8 : 0)][dcol];
    return bv;
  };

  const int nchunks = (q_base + BQ) / KC;  // keys 0 .. q_base+127 (causal)

  // ---- preload chunk 0 via Tensor Data Mover ----
  if (wave == 0) tdm_issue(kptr);

  for (int ck = 0; ck < nchunks; ++ck) {
    const int kc = ck * KC;

    // ---- publish TDM staging to the block ----
    if (wave == 0) __builtin_amdgcn_s_wait_tensorcnt(0);
    __syncthreads();

    // ---- convert staging f32 -> f16 K/V tiles (LDS -> LDS, 8 fl4/thread) ----
#pragma unroll
    for (int i = 0; i < 8; ++i) {
      const int idx = (tid + i * 256) * 4;      // 8192 floats total
      const int key = idx >> 8;                 // 256 floats per seq row
      const int rem = idx & 255;
      const int isv = rem >> 7;                 // 0=K plane, 1=V plane
      const int d   = rem & 127;
      const float4 x = *(const float4*)&Stg[key][isv][d];
      _Float16* dst = &KVs[isv][key][d];
      dst[0] = (_Float16)x.x; dst[1] = (_Float16)x.y;
      dst[2] = (_Float16)x.z; dst[3] = (_Float16)x.w;
    }
    __syncthreads();  // staging reusable; f16 tiles valid for everyone

    // ---- kick TDM for the next chunk; overlaps with compute below ----
    if (wave == 0 && ck + 1 < nchunks)
      tdm_issue(kptr + (size_t)(kc + KC) * STRIDE_S);

    // wave-uniform causal skip: chunk fully in the future for this wave
    if (kc <= q0 + QT - 1) {
      // ---- scores: S = Q K^T, two 16x16 tiles, pipelined B loads ----
      const _Float16* kr0 = &KVs[0][ln][0];
      const _Float16* kr1 = &KVs[0][16 + ln][0];
      v8f sc0 = vzero8(), sc1 = vzero8();
      v16h b0 = ldb(kr0, 0), b1 = ldb(kr1, 0);
#pragma unroll
      for (int c = 0; c < 4; ++c) {
        v16h n0 = b0, n1 = b1;
        if (c < 3) { n0 = ldb(kr0, c + 1); n1 = ldb(kr1, c + 1); }
        sc0 = __builtin_amdgcn_wmma_f32_16x16x32_f16(
            false, qa[c], false, b0, (short)0, sc0, false, false);
        sc1 = __builtin_amdgcn_wmma_f32_16x16x32_f16(
            false, qa[c], false, b1, (short)0, sc1, false, false);
        b0 = n0; b1 = n1;
      }

      // ---- causal mask + batched row-max butterflies ----
      float s0v[8], s1v[8], mx[8];
#pragma unroll
      for (int r = 0; r < 8; ++r) {
        const int row = q0 + hi * 8 + r;
        s0v[r] = (kc + ln      <= row) ? sc0[r] : -1e30f;
        s1v[r] = (kc + 16 + ln <= row) ? sc1[r] : -1e30f;
        mx[r]  = fmaxf(s0v[r], s1v[r]);
      }
      swz_max8<0x041F>(mx);   // xor 1
      swz_max8<0x081F>(mx);   // xor 2
      swz_max8<0x101F>(mx);   // xor 4
      swz_max8<0x201F>(mx);   // xor 8  -> row max over 16 lanes

      // ---- online softmax update (exp2 domain) + park P in LDS ----
      float alpha[8];
#pragma unroll
      for (int r = 0; r < 8; ++r) {
        const float mnew = fmaxf(mrow[r], mx[r]);
        const float p0 = __builtin_amdgcn_exp2f(s0v[r] - mnew);
        const float p1 = __builtin_amdgcn_exp2f(s1v[r] - mnew);
        alpha[r] = __builtin_amdgcn_exp2f(mrow[r] - mnew);
        mrow[r] = mnew;
        Ps[wave][hi * 8 + r][ln]      = (_Float16)p0;
        Ps[wave][hi * 8 + r][16 + ln] = (_Float16)p1;
      }
      asm volatile("s_wait_dscnt 0" ::: "memory");

      // ---- reload P in A-layout (16x32, K = 32 keys) ----
      v16h pa;
      {
        const _Float16* pr = &Ps[wave][ln][0];
#pragma unroll
        for (int e = 0; e < 8; ++e) {
          pa[e]     = pr[hi * 8 + e];
          pa[8 + e] = pr[16 + hi * 8 + e];
        }
      }

      // ---- row-sum of P with one WMMA ----
      v8f ssum = __builtin_amdgcn_wmma_f32_16x16x32_f16(
          false, pa, false, vones, (short)0, vzero8(), false, false);
#pragma unroll
      for (int r = 0; r < 8; ++r) lrow[r] = lrow[r] * alpha[r] + ssum[r];

      // ---- O = O*alpha + P V : 8 d-tiles, pipelined V-operand loads ----
      v16h bv = ldv(0);
#pragma unroll
      for (int dt = 0; dt < 8; ++dt) {
        v16h nv = bv;
        if (dt < 7) nv = ldv(dt + 1);
#pragma unroll
        for (int r = 0; r < 8; ++r) oacc[dt][r] *= alpha[r];
        oacc[dt] = __builtin_amdgcn_wmma_f32_16x16x32_f16(
            false, pa, false, bv, (short)0, oacc[dt], false, false);
        bv = nv;
      }
    }
    __syncthreads();  // f16 tiles consumed before next conversion overwrites
  }

  // ---- epilogue: normalize by row sum (fast rcp), store fp32 (B,S,H,D) ----
  float inv[8];
#pragma unroll
  for (int r = 0; r < 8; ++r) inv[r] = __builtin_amdgcn_rcpf(lrow[r]);
#pragma unroll
  for (int dt = 0; dt < 8; ++dt) {
    const int d = dt * 16 + ln;
#pragma unroll
    for (int r = 0; r < 8; ++r) {
      const int row = q0 + hi * 8 + r;
      out[((size_t)(b * Sc + row) * Hc + h) * Dc + d] = oacc[dt][r] * inv[r];
    }
  }
}

extern "C" void kernel_launch(void* const* d_in, const int* in_sizes, int n_in,
                              void* d_out, int out_size, void* d_ws, size_t ws_size,
                              hipStream_t stream) {
  (void)in_sizes; (void)n_in; (void)out_size; (void)d_ws; (void)ws_size;
  const float* qkv = (const float*)d_in[0];
  float* out = (float*)d_out;
  const int grid = Bc * Hc * (Sc / BQ);  // 1024 blocks
  fa_fwd_f16wmma<<<grid, NW * 32, 0, stream>>>(qkv, out);
}
